// ParallelCoAttentionNetwork_49392123904252
// MI455X (gfx1250) — matrix operand order repaired and checked by reading, not compiled, not run on hardware
//
#include <hip/hip_runtime.h>

// Problem dims (fixed by the reference)
#define BB 64
#define HH 1024
#define RR 196
#define RP 256   // R padded to a multiple of 64 (tile-exact, zero-filled)
#define LL 512
#define KK 512

typedef __attribute__((ext_vector_type(4)))  unsigned int   v4u32;
typedef __attribute__((ext_vector_type(8)))  unsigned int   v8u32;
typedef __attribute__((ext_vector_type(16))) __bf16         v16bf;
typedef __attribute__((ext_vector_type(8)))  float          v8f;

__device__ __forceinline__ unsigned short f2b(float f) {
    unsigned int u = __float_as_uint(f);
    u += 0x7FFFu + ((u >> 16) & 1u);          // round-to-nearest-even
    return (unsigned short)(u >> 16);
}
__device__ __forceinline__ float b2f(unsigned short s) {
    return __uint_as_float(((unsigned int)s) << 16);
}

// A fragment: lane -> M, regs -> K (ISA 16-bit A 16x32 layout); two 16B chunks
__device__ __forceinline__ v8u32 load_a16(const unsigned short* row, int k0, int hf) {
    uint4 lo = *(const uint4*)(row + k0 + 8 * hf);
    uint4 hi = *(const uint4*)(row + k0 + 16 + 8 * hf);
    v8u32 a;
    a[0] = lo.x; a[1] = lo.y; a[2] = lo.z; a[3] = lo.w;
    a[4] = hi.x; a[5] = hi.y; a[6] = hi.z; a[7] = hi.w;
    return a;
}
// B fragment from a k-major LDS tile (row stride 72 elems, 16B-aligned rows)
__device__ __forceinline__ v8u32 load_b_lds(const unsigned short* bt, int n, int hf) {
    v8u32 b;
#pragma unroll
    for (int i = 0; i < 8; ++i) {
        int k = (i < 4) ? (8 * hf + 2 * i) : (16 + 8 * hf + 2 * (i - 4));
        b[i] = (unsigned int)bt[k * 72 + n] | ((unsigned int)bt[(k + 1) * 72 + n] << 16);
    }
    return b;
}

__device__ __forceinline__ v8f wmma_bf16(const v8u32& a, const v8u32& b, const v8f& c) {
    return __builtin_amdgcn_wmma_f32_16x16x32_bf16(
        false, __builtin_bit_cast(v16bf, a), false, __builtin_bit_cast(v16bf, b),
        (short)0, c, false, false);
}

// ---------------------------------------------------------------------------
// TDM: DMA one [32 rows x 64 cols] bf16 tile (row stride ldb elems) into LDS.
// LDS rows are padded 128B -> 144B (pad_interval=32 DW, pad_amount=4 DW) to
// dodge bank conflicts on the 16-bit B fragment reads.
// D# layout per CDNA5 ISA (2D, groups 0+1 only).
// ---------------------------------------------------------------------------
__device__ __forceinline__ void tdm_load_b_tile(const unsigned short* gsrc,
                                                unsigned int lds_off, int ldb) {
    unsigned long long ga = (unsigned long long)(size_t)gsrc;
    v4u32 g0;
    g0[0] = 1u;                                               // count=1, user mode
    g0[1] = lds_off;                                          // lds_addr (bytes)
    g0[2] = (unsigned int)ga;                                 // global_addr[31:0]
    g0[3] = ((unsigned int)(ga >> 32) & 0x01FFFFFFu)          // global_addr[56:32]
          | 0x80000000u;                                      // type=2 ("image")
    unsigned int td0 = (unsigned int)ldb;                     // tensor_dim0
    v8u32 g1;
    g1[0] = 0x07110000u;   // data_size=2B | pad_enable | pad_interval=32DW | pad_amount=4DW
    g1[1] = (td0 & 0xFFFFu) << 16;                            // tensor_dim0 lo
    g1[2] = (32u << 16) | (td0 >> 16);                        // tensor_dim1=32 | dim0 hi
    g1[3] = (64u << 16);                                      // tile_dim0=64
    g1[4] = 32u;                                              // tile_dim1=32, tile_dim2=0
    g1[5] = td0;                                              // tensor_dim0_stride lo32
    g1[6] = 0u;                                               // stride hi / dim1_stride lo
    g1[7] = 0u;
    asm volatile("tensor_load_to_lds %0, %1" :: "s"(g0), "s"(g1) : "memory");
}

// ---------------------------------------------------------------------------
// f32 -> bf16 bulk conversion (no padding)
// ---------------------------------------------------------------------------
__global__ void cvt_bf16_kernel(const float* __restrict__ in,
                                unsigned short* __restrict__ out, long long n) {
    long long i = (long long)blockIdx.x * blockDim.x + threadIdx.x;
    if (i < n) out[i] = f2b(in[i]);
}

// f32 [rows,R] -> bf16 [rows,Rp], pad columns zero-filled
__global__ void cvt_pad_kernel(const float* __restrict__ in,
                               unsigned short* __restrict__ out,
                               long long rows, int R, int Rp) {
    long long i = (long long)blockIdx.x * blockDim.x + threadIdx.x;
    if (i >= rows * (long long)Rp) return;
    int r = (int)(i % Rp);
    long long row = i / Rp;
    out[i] = (r < R) ? f2b(in[row * R + r]) : (unsigned short)0;
}

// ---------------------------------------------------------------------------
// Shared GEMM epilogues (C/D layout: lane -> N, reg v -> M = v + 8*hf)
// ---------------------------------------------------------------------------
__device__ __forceinline__ void epi_store(unsigned short* Db, int ldd, int m0,
                                          int nA, int nB, int hf,
                                          const v8f& a00, const v8f& a01,
                                          const v8f& a10, const v8f& a11) {
#pragma unroll
    for (int v = 0; v < 8; ++v) {
        int m = m0 + v + 8 * hf;
        Db[(long long)m * ldd + nA]        = f2b(a00[v]);
        Db[(long long)m * ldd + nB]        = f2b(a01[v]);
        Db[(long long)(m + 16) * ldd + nA] = f2b(a10[v]);
        Db[(long long)(m + 16) * ldd + nB] = f2b(a11[v]);
    }
}
__device__ __forceinline__ void epi_fused(const unsigned short* Cb, int ldcin,
                                          float* Sred, const float* wvec,
                                          long long b, int m0, int nA, int nB,
                                          int hf, int Nval,
                                          const v8f& a00, const v8f& a01,
                                          const v8f& a10, const v8f& a11) {
    float l0 = 0.f, l1 = 0.f;
#pragma unroll
    for (int v = 0; v < 8; ++v) {
        int m = m0 + v + 8 * hf;
        float w0 = b2f(f2b(wvec[m]));          // bf16-rounded weight, like ref
        float w1 = b2f(f2b(wvec[m + 16]));
        l0 += w0 * tanhf(a00[v] + b2f(Cb[(long long)m * ldcin + nA]));
        l1 += w0 * tanhf(a01[v] + b2f(Cb[(long long)m * ldcin + nB]));
        l0 += w1 * tanhf(a10[v] + b2f(Cb[(long long)(m + 16) * ldcin + nA]));
        l1 += w1 * tanhf(a11[v] + b2f(Cb[(long long)(m + 16) * ldcin + nB]));
    }
    l0 += __shfl_xor(l0, 16, 32);              // combine the two M-halves
    l1 += __shfl_xor(l1, 16, 32);
    if (hf == 0) {
        if (nA < Nval) atomicAdd(&Sred[b * Nval + nA], l0);
        if (nB < Nval) atomicAdd(&Sred[b * Nval + nB], l1);
    }
}

// ---------------------------------------------------------------------------
// Non-transposed-B GEMM: B tiles staged by the Tensor Data Mover into LDS,
// double-buffered (TENSORcnt pipelining). 32x32 wave tile, 128x64 per block.
// ---------------------------------------------------------------------------
template <bool FUSED>
__global__ void gemm_wmma_nt_kernel(
    const unsigned short* __restrict__ A,   int lda,   long long strideA,
    const unsigned short* __restrict__ Bm,  int ldb,   long long strideB,
    const unsigned short* __restrict__ Cin, int ldcin, long long strideCin,
    unsigned short* __restrict__ D,         int ldd,   long long strideD,
    float* __restrict__ Sred, const float* __restrict__ wvec,
    int Kd, int Nval)
{
    __shared__ __align__(16) unsigned short Bs[2][32 * 72];   // 144B LDS rows (padded)
    const int lane = threadIdx.x & 31;
    const int wave = threadIdx.x >> 5;
    const int hf   = lane >> 4;
    const int ln16 = lane & 15;
    const int b    = blockIdx.z;
    const int nblk = blockIdx.x * 64;
    const int n0w  = (wave & 1) * 32;
    const int m0   = (blockIdx.y * 4 + (wave >> 1)) * 32;

    const unsigned short* Ab = A  + (long long)b * strideA;
    const unsigned short* Bb = Bm + (long long)b * strideB + nblk;
    const unsigned short* Arow0 = Ab + (long long)(m0 + ln16) * lda;
    const unsigned short* Arow1 = Arow0 + 16 * (long long)lda;

    v8f acc00, acc01, acc10, acc11;
#pragma unroll
    for (int i = 0; i < 8; ++i) { acc00[i] = 0.f; acc01[i] = 0.f; acc10[i] = 0.f; acc11[i] = 0.f; }

    const int steps = Kd >> 5;
    if (wave == 0)
        tdm_load_b_tile(Bb, (unsigned int)(size_t)&Bs[0][0], ldb);

    for (int s = 0; s < steps; ++s) {
        const int k0 = s << 5;
        if (wave == 0) {
            if (s + 1 < steps) {
                tdm_load_b_tile(Bb + (long long)(k0 + 32) * ldb,
                                (unsigned int)(size_t)&Bs[(s + 1) & 1][0], ldb);
                __builtin_amdgcn_s_wait_tensorcnt((short)1);   // current tile done
            } else {
                __builtin_amdgcn_s_wait_tensorcnt((short)0);
            }
        }
        __syncthreads();

        __builtin_prefetch(Arow0 + k0 + 64, 0, 1);   // global_prefetch_b8
        __builtin_prefetch(Arow1 + k0 + 64, 0, 1);
        v8u32 a0 = load_a16(Arow0, k0, hf);
        v8u32 a1 = load_a16(Arow1, k0, hf);

        const unsigned short* Brow = &Bs[s & 1][0];
        v8u32 b0 = load_b_lds(Brow, n0w + ln16, hf);
        v8u32 b1 = load_b_lds(Brow, n0w + 16 + ln16, hf);

        acc00 = wmma_bf16(a0, b0, acc00);
        acc01 = wmma_bf16(a0, b1, acc01);
        acc10 = wmma_bf16(a1, b0, acc10);
        acc11 = wmma_bf16(a1, b1, acc11);
        __syncthreads();
    }

    const int nA = nblk + n0w + ln16;
    const int nB = nA + 16;
    if (!FUSED) {
        epi_store(D + (long long)b * strideD, ldd, m0, nA, nB, hf, acc00, acc01, acc10, acc11);
    } else {
        epi_fused(Cin + (long long)b * strideCin, ldcin, Sred, wvec,
                  b, m0, nA, nB, hf, Nval, acc00, acc01, acc10, acc11);
    }
}

// ---------------------------------------------------------------------------
// Transposed-B GEMM: B element (k,n) = Bsrc[n*ldb + k]. Cooperative LDS
// transpose: coalesced global_load_b128 per thread + 8x ds_store_b16 scatter
// into a k-major padded tile, fragments then read with 16-bit LDS loads.
// ---------------------------------------------------------------------------
template <bool FUSED>
__global__ void gemm_wmma_bt_kernel(
    const unsigned short* __restrict__ A,   int lda,   long long strideA,
    const unsigned short* __restrict__ Bm,  int ldb,   long long strideB,
    const unsigned short* __restrict__ Cin, int ldcin, long long strideCin,
    unsigned short* __restrict__ D,         int ldd,   long long strideD,
    float* __restrict__ Sred, const float* __restrict__ wvec,
    int Kd, int Nval)
{
    __shared__ __align__(16) unsigned short Bs[32][72];       // k-major, padded rows
    const int tid  = threadIdx.x;
    const int lane = tid & 31;
    const int wave = tid >> 5;
    const int hf   = lane >> 4;
    const int ln16 = lane & 15;
    const int b    = blockIdx.z;
    const int nblk = blockIdx.x * 64;
    const int n0w  = (wave & 1) * 32;
    const int m0   = (blockIdx.y * 4 + (wave >> 1)) * 32;
    const int nst  = tid >> 2;        // 0..63: source row staged by this thread
    const int qk   = (tid & 3) * 8;   // k sub-offset within the 32-wide step

    const unsigned short* Ab = A  + (long long)b * strideA;
    const unsigned short* Bb = Bm + (long long)b * strideB;
    const unsigned short* Arow0 = Ab + (long long)(m0 + ln16) * lda;
    const unsigned short* Arow1 = Arow0 + 16 * (long long)lda;
    const unsigned short* Bst   = Bb + (long long)(nblk + nst) * ldb + qk;

    v8f acc00, acc01, acc10, acc11;
#pragma unroll
    for (int i = 0; i < 8; ++i) { acc00[i] = 0.f; acc01[i] = 0.f; acc10[i] = 0.f; acc11[i] = 0.f; }

    for (int k0 = 0; k0 < Kd; k0 += 32) {
        uint4 d = *(const uint4*)(Bst + k0);            // 8 contiguous K-elems of row nst
        __syncthreads();                                // prior reads of Bs complete
        unsigned int wlo[4] = {d.x, d.y, d.z, d.w};
#pragma unroll
        for (int e = 0; e < 4; ++e) {                   // transpose scatter
            Bs[qk + 2 * e][nst]     = (unsigned short)(wlo[e] & 0xFFFFu);
            Bs[qk + 2 * e + 1][nst] = (unsigned short)(wlo[e] >> 16);
        }
        __syncthreads();

        __builtin_prefetch(Arow0 + k0 + 64, 0, 1);
        __builtin_prefetch(Arow1 + k0 + 64, 0, 1);
        v8u32 a0 = load_a16(Arow0, k0, hf);
        v8u32 a1 = load_a16(Arow1, k0, hf);
        const unsigned short* Brow = &Bs[0][0];
        v8u32 b0 = load_b_lds(Brow, n0w + ln16, hf);
        v8u32 b1 = load_b_lds(Brow, n0w + 16 + ln16, hf);

        acc00 = wmma_bf16(a0, b0, acc00);
        acc01 = wmma_bf16(a0, b1, acc01);
        acc10 = wmma_bf16(a1, b0, acc10);
        acc11 = wmma_bf16(a1, b1, acc11);
    }

    const int nA = nblk + n0w + ln16;
    const int nB = nA + 16;
    if (!FUSED) {
        epi_store(D + (long long)b * strideD, ldd, m0, nA, nB, hf, acc00, acc01, acc10, acc11);
    } else {
        epi_fused(Cin + (long long)b * strideCin, ldcin, Sred, wvec,
                  b, m0, nA, nB, hf, Nval, acc00, acc01, acc10, acc11);
    }
}

// ---------------------------------------------------------------------------
// softmax over R=196 (one block per batch, 256 threads) -> a_v (bf16)
// ---------------------------------------------------------------------------
__global__ void softmax_v_kernel(const float* __restrict__ s,
                                 unsigned short* __restrict__ out, int R) {
    __shared__ float red[256];
    int b = blockIdx.x, t = threadIdx.x;
    float x = (t < R) ? s[(long long)b * R + t] : -3.0e38f;
    red[t] = x; __syncthreads();
    for (int o = 128; o > 0; o >>= 1) { if (t < o) red[t] = fmaxf(red[t], red[t + o]); __syncthreads(); }
    float mx = red[0]; __syncthreads();
    float e = (t < R) ? __expf(x - mx) : 0.0f;
    red[t] = e; __syncthreads();
    for (int o = 128; o > 0; o >>= 1) { if (t < o) red[t] += red[t + o]; __syncthreads(); }
    float sum = red[0];
    if (t < R) out[(long long)b * R + t] = f2b(e / sum);
}

// ---------------------------------------------------------------------------
// softmax over L=512 then masked re-softmax (one block per batch, 512 thr)
// ---------------------------------------------------------------------------
__global__ void softmax_q_kernel(const float* __restrict__ s,
                                 const long long* __restrict__ lens,
                                 unsigned short* __restrict__ out, int L) {
    __shared__ float red[512];
    int b = blockIdx.x, t = threadIdx.x;
    float x = s[(long long)b * L + t];
    red[t] = x; __syncthreads();
    for (int o = 256; o > 0; o >>= 1) { if (t < o) red[t] = fmaxf(red[t], red[t + o]); __syncthreads(); }
    float mx = red[0]; __syncthreads();
    float e = __expf(x - mx);
    red[t] = e; __syncthreads();
    for (int o = 256; o > 0; o >>= 1) { if (t < o) red[t] += red[t + o]; __syncthreads(); }
    float sum = red[0]; __syncthreads();
    float aq = b2f(f2b(e / sum));                 // bf16 round-trip like ref
    bool m = (long long)t < lens[b];
    float sc = m ? aq : -3.0e38f;
    red[t] = sc; __syncthreads();
    for (int o = 256; o > 0; o >>= 1) { if (t < o) red[t] = fmaxf(red[t], red[t + o]); __syncthreads(); }
    float mx2 = red[0]; __syncthreads();
    float e2 = m ? __expf(sc - mx2) : 0.0f;
    red[t] = e2; __syncthreads();
    for (int o = 256; o > 0; o >>= 1) { if (t < o) red[t] += red[t + o]; __syncthreads(); }
    float sum2 = red[0];
    out[(long long)b * L + t] = f2b(e2 / sum2);
}

// ---------------------------------------------------------------------------
// v[b,h] = sum_r a_v[b,r] * V[b,h,r]   (one wave per h; lanes stride over r)
// ---------------------------------------------------------------------------
__global__ void final_v_kernel(const unsigned short* __restrict__ av,
                               const unsigned short* __restrict__ Vb,
                               unsigned short* __restrict__ outv,
                               int H, int R, int Rp) {
    int lane = threadIdx.x & 31, wave = threadIdx.x >> 5;
    int b = blockIdx.y;
    int h = blockIdx.x * 8 + wave;
    const unsigned short* Vrow = Vb + ((long long)b * H + h) * Rp;
    const unsigned short* arow = av + (long long)b * R;
    float acc = 0.0f;
    for (int r = lane; r < R; r += 32) acc += b2f(arow[r]) * b2f(Vrow[r]);
#pragma unroll
    for (int o = 16; o > 0; o >>= 1) acc += __shfl_xor(acc, o, 32);
    if (lane == 0) outv[(long long)b * H + h] = f2b(acc);
}

// ---------------------------------------------------------------------------
// q[b,h] = sum_l maq[b,l] * Q[b,l,h]  (threads over h -> coalesced Q reads)
// ---------------------------------------------------------------------------
__global__ void final_q_kernel(const unsigned short* __restrict__ maq,
                               const unsigned short* __restrict__ Qb,
                               unsigned short* __restrict__ outq, int H, int L) {
    __shared__ float a[LL];
    int t = threadIdx.x, b = blockIdx.y;
    for (int l = t; l < L; l += 256) a[l] = b2f(maq[(long long)b * L + l]);
    __syncthreads();
    int h = blockIdx.x * 256 + t;
    const unsigned short* Qbb = Qb + (long long)b * L * H;
    float acc = 0.0f;
    for (int l = 0; l < L; ++l) acc += a[l] * b2f(Qbb[(long long)l * H + h]);
    outq[(long long)b * H + h] = f2b(acc);
}

// ---------------------------------------------------------------------------
extern "C" void kernel_launch(void* const* d_in, const int* in_sizes, int n_in,
                              void* d_out, int out_size, void* d_ws, size_t ws_size,
                              hipStream_t stream) {
    (void)in_sizes; (void)n_in; (void)out_size; (void)ws_size;
    const int B = BB, H = HH, R = RR, Rp = RP, L = LL, K = KK;

    const float*     V    = (const float*)d_in[0];
    const float*     Q    = (const float*)d_in[1];
    const long long* Qlen = (const long long*)d_in[2];
    const float*     W_b  = (const float*)d_in[3];
    const float*     W_v  = (const float*)d_in[4];
    const float*     W_q  = (const float*)d_in[5];
    const float*     w_hv = (const float*)d_in[6];
    const float*     w_hq = (const float*)d_in[7];

    // ---- workspace carve-out (256B aligned) ----
    char* w = (char*)d_ws; size_t o = 0;
    auto take = [&](size_t bytes) { void* p = w + o; o = (o + bytes + 255) & ~(size_t)255; return p; };
    const long long nVp = (long long)B * H * Rp;   // padded V / WbV
    const long long nQ  = (long long)B * L * H;
    unsigned short* Vb    = (unsigned short*)take(nVp * 2);
    unsigned short* Qb    = (unsigned short*)take(nQ * 2);
    unsigned short* Wbb   = (unsigned short*)take((size_t)H * H * 2);
    unsigned short* Wvb   = (unsigned short*)take((size_t)K * H * 2);
    unsigned short* Wqb   = (unsigned short*)take((size_t)K * H * 2);
    unsigned short* WbVb  = (unsigned short*)take(nVp * 2);                    // [B,H,Rp]
    unsigned short* Cb    = (unsigned short*)take((size_t)B * L * Rp * 2);     // [B,L,Rp]
    unsigned short* WvVb  = (unsigned short*)take((size_t)B * K * Rp * 2);     // [B,K,Rp]
    unsigned short* WqQtb = (unsigned short*)take((size_t)B * K * L * 2);      // [B,K,L]
    float*          s_v   = (float*)take((size_t)B * R * 4);
    float*          s_q   = (float*)take((size_t)B * L * 4);

    // ---- d_out partition: a_v | masked_a_q | v | q (all bf16) ----
    unsigned short* out    = (unsigned short*)d_out;
    unsigned short* av_out = out;
    unsigned short* mq_out = out + (size_t)B * R;
    unsigned short* v_out  = mq_out + (size_t)B * L;
    unsigned short* q_out  = v_out + (size_t)B * H;

    // ---- 1. convert inputs to bf16 (V padded to Rp with zeros) ----
    cvt_pad_kernel<<<(unsigned)((nVp + 255) / 256), 256, 0, stream>>>(V, Vb, (long long)B * H, R, Rp);
    cvt_bf16_kernel<<<(unsigned)((nQ + 255) / 256), 256, 0, stream>>>(Q, Qb, nQ);
    cvt_bf16_kernel<<<(H * H) / 256, 256, 0, stream>>>(W_b, Wbb, (long long)H * H);
    cvt_bf16_kernel<<<(K * H) / 256, 256, 0, stream>>>(W_v, Wvb, (long long)K * H);
    cvt_bf16_kernel<<<(K * H) / 256, 256, 0, stream>>>(W_q, Wqb, (long long)K * H);

    hipMemsetAsync(s_v, 0, (size_t)B * R * 4, stream);
    hipMemsetAsync(s_q, 0, (size_t)B * L * 4, stream);

    // ---- 2. WbV[b] = Wb @ V[b]          [H,Rp], Kd=H ----
    gemm_wmma_nt_kernel<false><<<dim3(Rp / 64, H / 128, B), 256, 0, stream>>>(
        Wbb, H, 0, Vb, Rp, (long long)H * Rp, nullptr, 0, 0,
        WbVb, Rp, (long long)H * Rp, nullptr, nullptr, H, Rp);
    // ---- 3. C[b] = Q[b] @ WbV[b]        [L,Rp], Kd=H ----
    gemm_wmma_nt_kernel<false><<<dim3(Rp / 64, L / 128, B), 256, 0, stream>>>(
        Qb, H, (long long)L * H, WbVb, Rp, (long long)H * Rp, nullptr, 0, 0,
        Cb, Rp, (long long)L * Rp, nullptr, nullptr, H, Rp);
    // ---- 4. WvV[b] = Wv @ V[b]          [K,Rp], Kd=H ----
    gemm_wmma_nt_kernel<false><<<dim3(Rp / 64, K / 128, B), 256, 0, stream>>>(
        Wvb, H, 0, Vb, Rp, (long long)H * Rp, nullptr, 0, 0,
        WvVb, Rp, (long long)K * Rp, nullptr, nullptr, H, Rp);
    // ---- 5. WqQt[b] = Wq @ Q[b]^T       [K,L], Kd=H (B transposed access) ----
    gemm_wmma_bt_kernel<false><<<dim3(L / 64, K / 128, B), 256, 0, stream>>>(
        Wqb, H, 0, Qb, H, (long long)L * H, nullptr, 0, 0,
        WqQtb, L, (long long)K * L, nullptr, nullptr, H, L);
    // ---- 6. s_v[b,r] = sum_k w_hv[k]*tanh(WvV + WqQt@C)   Kd=L (fused) ----
    gemm_wmma_nt_kernel<true><<<dim3(Rp / 64, K / 128, B), 256, 0, stream>>>(
        WqQtb, L, (long long)K * L, Cb, Rp, (long long)L * Rp,
        WvVb, Rp, (long long)K * Rp, nullptr, 0, 0,
        s_v, w_hv, L, R);
    // ---- 7. s_q[b,l] = sum_k w_hq[k]*tanh(WqQt + WvV@C^T) Kd=Rp (fused) ----
    gemm_wmma_bt_kernel<true><<<dim3(L / 64, K / 128, B), 256, 0, stream>>>(
        WvVb, Rp, (long long)K * Rp, Cb, Rp, (long long)L * Rp,
        WqQtb, L, (long long)K * L, nullptr, 0, 0,
        s_q, w_hq, Rp, L);

    // ---- 8. softmaxes ----
    softmax_v_kernel<<<B, 256, 0, stream>>>(s_v, av_out, R);
    softmax_q_kernel<<<B, 512, 0, stream>>>(s_q, Qlen, mq_out, L);

    // ---- 9. final matrix-vector products ----
    final_v_kernel<<<dim3(H / 8, B), 256, 0, stream>>>(av_out, Vb, v_out, H, R, Rp);
    final_q_kernel<<<dim3(H / 256, B), 256, 0, stream>>>(mq_out, Qb, q_out, H, L);
}
